// GAT_17162689314843
// MI455X (gfx1250) — matrix operand order repaired and checked by reading, not compiled, not run on hardware
//
#include <hip/hip_runtime.h>
#include <hip/hip_bf16.h>

typedef _Float16 half_t;
typedef __attribute__((ext_vector_type(16))) _Float16 v16h;
typedef __attribute__((ext_vector_type(8)))  _Float16 v8h;
typedef __attribute__((ext_vector_type(8)))  float    v8f;

#define LEAKY_SLOPE 0.2f

union V16 { v16h v; v8h h[2]; };

// ---------- helpers ----------
__device__ __forceinline__ unsigned fenc(float x) {
    unsigned u = __float_as_uint(x);
    return (u & 0x80000000u) ? ~u : (u | 0x80000000u);
}
__device__ __forceinline__ float fdec(unsigned v) {
    return __uint_as_float((v & 0x80000000u) ? (v & 0x7fffffffu) : ~v);
}

// ---------- utility kernels ----------
__global__ void k_cast_f16(const float* __restrict__ in, half_t* __restrict__ out, long long n) {
    long long i = blockIdx.x * (long long)blockDim.x + threadIdx.x;
    if (i < n) out[i] = (half_t)in[i];
}

__global__ void k_fill_f32(float* __restrict__ p, float v, long long n) {
    long long i = blockIdx.x * (long long)blockDim.x + threadIdx.x;
    if (i < n) p[i] = v;
}

__global__ void k_fill_u32(unsigned* __restrict__ p, unsigned v, long long n) {
    long long i = blockIdx.x * (long long)blockDim.x + threadIdx.x;
    if (i < n) p[i] = v;
}

// ---------- pack B (f32, row-major K x Nout) into WMMA operand layout (f16) ----------
// Per 32(K) x 16(N) tile: lane l keeps its 16 operand halves contiguous, so the
// GEMM reads one aligned 32-byte chunk per lane per wmma.
// Element (tile t = kt*(Nout/16)+nt16, lane, e):
//   k = kt*32 + (lane>=16 ? 16 : 0) + e ; col = nt16*16 + (lane&15)
__global__ void k_pack_b(const float* __restrict__ B, half_t* __restrict__ Bp, int K, int Nout) {
    long long i = blockIdx.x * (long long)blockDim.x + threadIdx.x;
    const long long total = (long long)K * Nout;
    if (i >= total) return;
    const int e    = (int)(i & 15);
    const int lane = (int)((i >> 4) & 31);
    const long long t = i >> 9;                // 32*16 = 512 elems per tile
    const int ntiles = Nout >> 4;
    const int kt   = (int)(t / ntiles);
    const int nt16 = (int)(t % ntiles);
    const int lo = lane & 15, hi = lane >> 4;
    const int k   = kt * 32 + hi * 16 + e;
    const int col = nt16 * 16 + lo;
    Bp[i] = (half_t)B[(long long)k * Nout + col];
}

// ---------- WMMA GEMM: C[M x Nout] = A[M x K] * Bp, f16 in / f32 out ----------
// A: row-major f16. Bp: packed by k_pack_b. One wave computes 16(M) x 64(N).
__global__ void k_gemm_wmma(const half_t* __restrict__ A, const half_t* __restrict__ Bp,
                            float* __restrict__ C, int M, int K, int Nout) {
    const int lane = threadIdx.x & 31;
    const int wave = threadIdx.x >> 5;
    const int gw   = blockIdx.x * (blockDim.x >> 5) + wave;
    const int stripsN = Nout >> 6;             // 64-wide strips
    const int mt = gw / stripsN;
    const int nt = gw % stripsN;
    if (mt * 16 >= M) return;                  // wave-uniform guard (EXEC stays all-1s)

    const int lo = lane & 15;
    const int hi = lane >> 4;
    const int ntiles16 = Nout >> 4;

    v8f acc[4] = {v8f{}, v8f{}, v8f{}, v8f{}};

    const long long arow = (long long)(mt * 16 + lo) * K;
    for (int k0 = 0; k0 < K; k0 += 32) {
        // A operand: lane lo = row M; element e -> K = hi*16 + e
        V16 a;
        a.h[0] = *(const v8h*)(A + arow + k0 + hi * 16);
        a.h[1] = *(const v8h*)(A + arow + k0 + hi * 16 + 8);

#pragma unroll
        for (int j = 0; j < 4; ++j) {
            const long long t = (long long)(k0 >> 5) * ntiles16 + (nt << 2) + j;
            const v16h b = *(const v16h*)(Bp + ((t << 5) + lane) * 16);
            acc[j] = __builtin_amdgcn_wmma_f32_16x16x32_f16(
                false, a.v, false, b, (short)0, acc[j], false, false);
        }
    }

    // C/D layout: VGPR v -> row = mt*16 + hi*8 + v, col = lane%16
#pragma unroll
    for (int j = 0; j < 4; ++j) {
        const int col = (nt << 6) + (j << 4) + lo;
#pragma unroll
        for (int v = 0; v < 8; ++v) {
            const int row = mt * 16 + hi * 8 + v;
            C[(long long)row * Nout + col] = acc[j][v];
        }
    }
}

// ---------- attention projections: el/er[n,h] = dot(h[n,h,:], attn_{l,r}[h,:]) ----------
__global__ void k_attn_scores(const float* __restrict__ h,
                              const float* __restrict__ attl, const float* __restrict__ attr,
                              float* __restrict__ el, float* __restrict__ er,
                              int Nn, int H, int F) {
    long long i = blockIdx.x * (long long)blockDim.x + threadIdx.x;
    if (i >= (long long)Nn * H) return;
    const int hd = (int)(i % H);
    const float* row = h + i * F;
    const float* al  = attl + hd * F;
    const float* ar  = attr + hd * F;
    float sl = 0.f, sr = 0.f;
    for (int c = 0; c < F; ++c) { float v = row[c]; sl += v * al[c]; sr += v * ar[c]; }
    el[i] = sl; er[i] = sr;
}

// ---------- edge pass A: e = leaky(el[src]+er[dst]); segment max via encoded atomicMax ----------
__global__ void k_edge_scores(const int* __restrict__ src, const int* __restrict__ dst,
                              const float* __restrict__ el, const float* __restrict__ er,
                              float* __restrict__ ebuf, unsigned* __restrict__ menc,
                              long long E, int H) {
    long long i = blockIdx.x * (long long)blockDim.x + threadIdx.x;
    if (i >= E * H) return;
    const long long e = i / H;
    const int hd = (int)(i % H);
    const int s = src[e], d = dst[e];
    float z = el[(long long)s * H + hd] + er[(long long)d * H + hd];
    z = z > 0.f ? z : LEAKY_SLOPE * z;
    ebuf[i] = z;
    atomicMax(&menc[(long long)d * H + hd], fenc(z));
}

// ---------- edge pass B: a = exp(e - m[dst]); segment sum ----------
__global__ void k_edge_norm(const int* __restrict__ dst,
                            float* __restrict__ ebuf, const unsigned* __restrict__ menc,
                            float* __restrict__ den, long long E, int H) {
    long long i = blockIdx.x * (long long)blockDim.x + threadIdx.x;
    if (i >= E * H) return;
    const long long e = i / H;
    const int hd = (int)(i % H);
    const int d = dst[e];
    const float m = fdec(menc[(long long)d * H + hd]);
    const float a = __expf(ebuf[i] - m);
    ebuf[i] = a;
    atomicAdd(&den[(long long)d * H + hd], a);
}

// ---------- edge pass C: out[dst] += h[src] * (a / denom[dst]) ----------
// one thread per (edge, float4-chunk); Ftot = H*F
__global__ void k_message(const int* __restrict__ src, const int* __restrict__ dst,
                          const float* __restrict__ h, const float* __restrict__ ebuf,
                          const float* __restrict__ den, float* __restrict__ out,
                          long long E, int H, int F, int Ftot) {
    const int chunks = Ftot >> 2;
    long long i = blockIdx.x * (long long)blockDim.x + threadIdx.x;
    if (i >= E * chunks) return;
    const int  q  = (int)(i % chunks);
    const long long e = i / chunks;
    const int  c0 = q << 2;
    const int  hd = c0 / F;
    const int  s  = src[e], d = dst[e];
    const float w = ebuf[e * H + hd] / den[(long long)d * H + hd];
    const float4 v = *(const float4*)(h + (long long)s * Ftot + c0);
    float* o = out + (long long)d * Ftot + c0;
    atomicAdd(o + 0, v.x * w);
    atomicAdd(o + 1, v.y * w);
    atomicAdd(o + 2, v.z * w);
    atomicAdd(o + 3, v.w * w);
}

// ---------- layer-1 combine: x2 = f16(elu(rst + x_identity + bias)) ----------
__global__ void k_combine1(const float* __restrict__ rst, const float* __restrict__ x,
                           const float* __restrict__ bias, half_t* __restrict__ xout,
                           long long total, int Ftot) {
    long long i = blockIdx.x * (long long)blockDim.x + threadIdx.x;
    if (i >= total) return;
    float v = rst[i] + x[i] + bias[(int)(i % Ftot)];
    v = v > 0.f ? v : (__expf(v) - 1.0f);     // ELU
    xout[i] = (half_t)v;
}

// ---------- add bias in place (layer 2 output pre-message init) ----------
__global__ void k_bias_add(float* __restrict__ out, const float* __restrict__ bias,
                           long long total, int Ftot) {
    long long i = blockIdx.x * (long long)blockDim.x + threadIdx.x;
    if (i >= total) return;
    out[i] += bias[(int)(i % Ftot)];
}

// =====================================================================
extern "C" void kernel_launch(void* const* d_in, const int* in_sizes, int n_in,
                              void* d_out, int out_size, void* d_ws, size_t ws_size,
                              hipStream_t stream) {
    const float* feat = (const float*)d_in[0];
    const int*   src  = (const int*)d_in[1];
    const int*   dst  = (const int*)d_in[2];
    const float* W1   = (const float*)d_in[3];
    const float* al1  = (const float*)d_in[4];
    const float* ar1  = (const float*)d_in[5];
    const float* b1   = (const float*)d_in[6];
    const float* W2   = (const float*)d_in[7];
    const float* al2  = (const float*)d_in[8];
    const float* ar2  = (const float*)d_in[9];
    const float* b2   = (const float*)d_in[10];
    const float* rW2  = (const float*)d_in[11];

    const int IN = 256, H1 = 8, HID = 32, OUT = 64;
    const int Nn = in_sizes[0] / IN;          // 50000
    const long long E = in_sizes[1];          // 850000

    // ---- carve scratch ----
    char* ws = (char*)d_ws;
    size_t off = 0;
    auto alloc = [&](size_t bytes) -> void* {
        void* p = ws + off;
        off = (off + bytes + 255) & ~(size_t)255;
        return p;
    };
    half_t*   xh   = (half_t*)alloc((size_t)Nn * IN * sizeof(half_t)); // f16 features; reused as x2h
    half_t*   w1p  = (half_t*)alloc((size_t)IN * IN * sizeof(half_t));   // packed W1
    half_t*   w2p  = (half_t*)alloc((size_t)IN * OUT * sizeof(half_t));  // packed W2
    half_t*   rw2p = (half_t*)alloc((size_t)IN * OUT * sizeof(half_t));  // packed res_W2
    float*    h1   = (float*)alloc((size_t)Nn * IN * sizeof(float));
    float*    rst1 = (float*)alloc((size_t)Nn * IN * sizeof(float));
    float*    el1  = (float*)alloc((size_t)Nn * H1 * sizeof(float));
    float*    er1  = (float*)alloc((size_t)Nn * H1 * sizeof(float));
    unsigned* m1   = (unsigned*)alloc((size_t)Nn * H1 * sizeof(unsigned));
    float*    dn1  = (float*)alloc((size_t)Nn * H1 * sizeof(float));
    float*    e1   = (float*)alloc((size_t)E * H1 * sizeof(float));
    float*    h2   = (float*)alloc((size_t)Nn * OUT * sizeof(float));
    float*    el2  = (float*)alloc((size_t)Nn * sizeof(float));
    float*    er2  = (float*)alloc((size_t)Nn * sizeof(float));
    unsigned* m2   = (unsigned*)alloc((size_t)Nn * sizeof(unsigned));
    float*    dn2  = (float*)alloc((size_t)Nn * sizeof(float));
    float*    e2   = (float*)alloc((size_t)E * sizeof(float));
    float*    out  = (float*)d_out;

    const int BS = 256;
    auto nb = [](long long n, int b) { return (unsigned)((n + b - 1) / b); };

    // ---- operand prep: cast A to f16; pack+cast B matrices to WMMA layout ----
    k_cast_f16<<<nb((long long)Nn * IN, BS), BS, 0, stream>>>(feat, xh, (long long)Nn * IN);
    k_pack_b<<<nb((long long)IN * IN, BS), BS, 0, stream>>>(W1, w1p, IN, IN);
    k_pack_b<<<nb((long long)IN * OUT, BS), BS, 0, stream>>>(W2, w2p, IN, OUT);
    k_pack_b<<<nb((long long)IN * OUT, BS), BS, 0, stream>>>(rW2, rw2p, IN, OUT);

    // ---- layer 1: h1 = X @ W1  (WMMA) ----
    {
        const int waves = ((Nn + 15) / 16) * (IN / 64);
        k_gemm_wmma<<<nb((long long)waves * 32, BS), BS, 0, stream>>>(xh, w1p, h1, Nn, IN, IN);
    }
    k_attn_scores<<<nb((long long)Nn * H1, BS), BS, 0, stream>>>(h1, al1, ar1, el1, er1, Nn, H1, HID);

    k_fill_u32<<<nb((long long)Nn * H1, BS), BS, 0, stream>>>(m1, 0u, (long long)Nn * H1);
    k_fill_f32<<<nb((long long)Nn * H1, BS), BS, 0, stream>>>(dn1, 0.f, (long long)Nn * H1);
    k_fill_f32<<<nb((long long)Nn * IN, BS), BS, 0, stream>>>(rst1, 0.f, (long long)Nn * IN);

    k_edge_scores<<<nb(E * H1, BS), BS, 0, stream>>>(src, dst, el1, er1, e1, m1, E, H1);
    k_edge_norm<<<nb(E * H1, BS), BS, 0, stream>>>(dst, e1, m1, dn1, E, H1);
    k_message<<<nb(E * (IN / 4), BS), BS, 0, stream>>>(src, dst, h1, e1, dn1, rst1, E, H1, HID, IN);

    // ---- combine: x2 = f16(elu(rst1 + feat + bias1)) (identity residual; reuse xh) ----
    k_combine1<<<nb((long long)Nn * IN, BS), BS, 0, stream>>>(rst1, feat, b1, xh, (long long)Nn * IN, IN);

    // ---- layer 2: h2 = X2 @ W2, residual straight into d_out (WMMA) ----
    {
        const int waves = ((Nn + 15) / 16) * (OUT / 64);
        k_gemm_wmma<<<nb((long long)waves * 32, BS), BS, 0, stream>>>(xh, w2p, h2, Nn, IN, OUT);
        k_gemm_wmma<<<nb((long long)waves * 32, BS), BS, 0, stream>>>(xh, rw2p, out, Nn, IN, OUT);
    }
    k_attn_scores<<<nb((long long)Nn, BS), BS, 0, stream>>>(h2, al2, ar2, el2, er2, Nn, 1, OUT);
    k_bias_add<<<nb((long long)Nn * OUT, BS), BS, 0, stream>>>(out, b2, (long long)Nn * OUT, OUT);

    k_fill_u32<<<nb((long long)Nn, BS), BS, 0, stream>>>(m2, 0u, (long long)Nn);
    k_fill_f32<<<nb((long long)Nn, BS), BS, 0, stream>>>(dn2, 0.f, (long long)Nn);

    k_edge_scores<<<nb(E, BS), BS, 0, stream>>>(src, dst, el2, er2, e2, m2, E, 1);
    k_edge_norm<<<nb(E, BS), BS, 0, stream>>>(dst, e2, m2, dn2, E, 1);
    k_message<<<nb(E * (OUT / 4), BS), BS, 0, stream>>>(src, dst, h2, e2, dn2, out, E, 1, OUT, OUT);
}
